// ELECT_Mnist_43894565765702
// MI455X (gfx1250) — compile-verified, matrix operand order checked by reading, not run on hardware
//
#include <hip/hip_runtime.h>
#include <hip/hip_bf16.h>
#include <math.h>

// Problem constants (from the reference)
#define GG   4096
#define NPG  75
#define EH   300
#define DD   128
#define H1   16
#define NL   4
#define NN   (GG * NPG)        // 307200 nodes
#define EE   (GG * 2 * EH)     // 2457600 edges
#define BN_EPS 1e-5f
#define LEAKY 0.01f
#define INV_SQRT75 0.11547005383792516f

typedef __attribute__((ext_vector_type(16))) _Float16 v16h;
typedef __attribute__((ext_vector_type(8)))  _Float16 v8h;
typedef __attribute__((ext_vector_type(8)))  float    v8f;

union F16Frag { v16h v; v8h h[2]; };

// Map (k within 0..127) -> (ks, hi, elem16) per the CDNA5 16-bit A/B fragment
// layout for V_WMMA_*_16X16X32: lanes 0-15 hold K in {ks*32+hi*8 .. +7} u
// {ks*32+hi*8+16 .. +23} packed as half-pairs along the 8 VGPRs.
__device__ __forceinline__ void frag_map(int k, int& ks, int& hi, int& elem16)
{
    ks = k >> 5;
    int r = k & 31;
    hi = (r >> 3) & 1;
    int base = r - 8 * hi;            // in {0..7} u {16..23}
    int j = (base < 8) ? (base >> 1) : (4 + ((base - 16) >> 1));
    elem16 = 2 * j + (base & 1);
}

// ---------------------------------------------------------------------------
// One-time weight pack: W (128x128 f32, row-major k*128+n) -> f16 fragments.
// Layout: [(ks*8 + w)*2 + h][lane][8 halves], h = lo/hi 16B chunk, so each
// (ks,w,h) chunk is a contiguous 32-lane x 16B block (conflict-free b128).
// ---------------------------------------------------------------------------
__global__ __launch_bounds__(256) void k_packW(const float* __restrict__ W,
                                               _Float16* __restrict__ Wp)
{
    int i = blockIdx.x * 256 + threadIdx.x;   // 16384 elements
    if (i >= 128 * 128) return;
    int k = i >> 7, n = i & 127;
    int ks, hi, e16;
    frag_map(k, ks, hi, e16);
    int h = e16 >> 3, e = e16 & 7;
    int lane = (n & 15) + 16 * hi;
    int w = n >> 4;
    Wp[(((ks * 8 + w) * 2 + h) * 32 + lane) * 8 + e] = (_Float16)W[i];
}

// ---------------------------------------------------------------------------
// WMMA GEMM: out = relu(A @ W + bias), A: n_rows x 128 (f32), Wp pre-packed.
// Block = 256 threads (8 waves); 16 rows per block, wave w -> cols [16w,16w+16).
// Inner loop: 4 x (2+2 ds_load_b128 + v_wmma_f32_16x16x32_f16).
// ---------------------------------------------------------------------------
__global__ __launch_bounds__(256) void k_gemm128_wmma(
    const float* __restrict__ A, const _Float16* __restrict__ Wp,
    const float* __restrict__ bias, float* __restrict__ out)
{
    __shared__ _Float16 Bs[16384];   // 32 KB packed weight fragments
    __shared__ _Float16 As[2048];    // 4 KB packed A fragments

    const int tid  = threadIdx.x;
    const int wave = tid >> 5;
    const int lane = tid & 31;
    const long m0  = (long)blockIdx.x * 16;

    // stage packed W: straight 16B copies (hot in L2)
    {
        const uint4* src = (const uint4*)Wp;
        uint4* dst = (uint4*)Bs;
        for (int i = tid; i < 2048; i += 256) dst[i] = src[i];
    }
    // stage A tile directly into fragment order (f32 -> f16 here)
    for (int i = tid; i < 16 * 128; i += 256) {
        int m = i >> 7, k = i & 127;
        int ks, hi, e16;
        frag_map(k, ks, hi, e16);
        int h = e16 >> 3, e = e16 & 7;
        int lane_a = m + 16 * hi;     // m in [0,16)
        As[((ks * 2 + h) * 32 + lane_a) * 8 + e] = (_Float16)A[(m0 + m) * 128 + k];
    }
    __syncthreads();

    const int col = wave * 16 + (lane & 15);
    const int hi  = lane >> 4;

    v8f acc = {};
#pragma unroll
    for (int ks = 0; ks < 4; ++ks) {
        F16Frag a, b;
        a.h[0] = *(const v8h*)&As[((ks * 2 + 0) * 32 + lane) * 8];
        a.h[1] = *(const v8h*)&As[((ks * 2 + 1) * 32 + lane) * 8];
        b.h[0] = *(const v8h*)&Bs[(((ks * 8 + wave) * 2 + 0) * 32 + lane) * 8];
        b.h[1] = *(const v8h*)&Bs[(((ks * 8 + wave) * 2 + 1) * 32 + lane) * 8];
        acc = __builtin_amdgcn_wmma_f32_16x16x32_f16(
            false, a.v, false, b.v, (short)0, acc, false, false);
    }

    const float bj = bias[col];
#pragma unroll
    for (int r = 0; r < 8; ++r) {
        long row = m0 + r + 8 * hi;
        float v = acc[r] + bj;
        out[row * 128 + col] = v > 0.0f ? v : 0.0f;
    }
}

// ---------------------------------------------------------------------------
// Edge aggregation (scalar, conv1): agg[col[e]] += x[row[e]]
// ---------------------------------------------------------------------------
__global__ void k_agg_scalar(const float* __restrict__ x, const int* __restrict__ row,
                             const int* __restrict__ col, float* __restrict__ aggs, int ne)
{
    int e = blockIdx.x * 256 + threadIdx.x;
    if (e >= ne) return;
    atomicAdd(&aggs[col[e]], x[row[e]]);
}

// Edge aggregation (vector, D=128): agg[col[e]][d] += h[row[e]][d]
// h fits in the 192MB L2, so these stay on-chip.
__global__ void k_agg_vec(const float* __restrict__ h, const int* __restrict__ row,
                          const int* __restrict__ col, float* __restrict__ agg, long total)
{
    long t = (long)blockIdx.x * 256 + threadIdx.x;
    if (t >= total) return;
    int e = (int)(t >> 7), d = (int)(t & 127);
    atomicAdd(&agg[(long)col[e] * 128 + d], h[(long)row[e] * 128 + d]);
}

// conv1 pre+expand: out[n][d] = relu(((1+eps)*x[n] + aggs[n]) * W1[d] + b1[d])
__global__ void k_conv1_expand(const float* __restrict__ x, const float* __restrict__ aggs,
                               const float* __restrict__ epsp, const float* __restrict__ W1,
                               const float* __restrict__ b1, float* __restrict__ out, long total)
{
    long t = (long)blockIdx.x * 256 + threadIdx.x;
    if (t >= total) return;
    int n = (int)(t >> 7), d = (int)(t & 127);
    float z = (1.0f + epsp[0]) * x[n] + aggs[n];
    float v = z * W1[d] + b1[d];
    out[t] = v > 0.0f ? v : 0.0f;
}

// agg += (1+eps)*h
__global__ void k_axpy_eps(float* __restrict__ agg, const float* __restrict__ h,
                           const float* __restrict__ epsp, long total)
{
    long t = (long)blockIdx.x * 256 + threadIdx.x;
    if (t >= total) return;
    agg[t] += (1.0f + epsp[0]) * h[t];
}

// ---------------------------------------------------------------------------
// BatchNorm: two-stage stats then apply
// ---------------------------------------------------------------------------
__global__ __launch_bounds__(128) void k_bn_stats(const float* __restrict__ h,
                                                  float* __restrict__ sum, float* __restrict__ sumsq,
                                                  int rows_per_block, long n_rows)
{
    int d = threadIdx.x;
    long r0 = (long)blockIdx.x * rows_per_block;
    float s = 0.0f, q = 0.0f;
    for (int r = 0; r < rows_per_block; ++r) {
        long row = r0 + r;
        if (row >= n_rows) break;
        float v = h[row * 128 + d];
        s += v; q += v * v;
    }
    atomicAdd(&sum[d], s);
    atomicAdd(&sumsq[d], q);
}

__global__ void k_bn_apply(const float* __restrict__ in, float* __restrict__ out,
                           const float* __restrict__ sum, const float* __restrict__ sumsq,
                           const float* __restrict__ g, const float* __restrict__ b,
                           int leaky, long total, float invN)
{
    long t = (long)blockIdx.x * 256 + threadIdx.x;
    if (t >= total) return;
    int d = (int)(t & 127);
    float m = sum[d] * invN;
    float var = sumsq[d] * invN - m * m;
    float rstd = rsqrtf(var + BN_EPS);
    float v = (in[t] - m) * rstd * g[d] + b[d];
    if (leaky) v = v > 0.0f ? v : LEAKY * v;
    out[t] = v;
}

// h = (residual ? h+z : z) * (1/sqrt(75))
__global__ void k_resscale(float* __restrict__ h, const float* __restrict__ z,
                           int residual, long total)
{
    long t = (long)blockIdx.x * 256 + threadIdx.x;
    if (t >= total) return;
    float v = residual ? (h[t] + z[t]) : z[t];
    h[t] = v * INV_SQRT75;
}

// ---------------------------------------------------------------------------
// Head: s[n] = leaky(leaky(h@W1+b1) @ W2 + b2)   (D=128 -> 16 -> 1)
// ---------------------------------------------------------------------------
__global__ void k_lin(const float* __restrict__ h, const float* __restrict__ W1,
                      const float* __restrict__ b1, const float* __restrict__ W2,
                      const float* __restrict__ b2, float* __restrict__ s, int n)
{
    int i = blockIdx.x * 256 + threadIdx.x;
    if (i >= n) return;
    const float* hr = h + (long)i * 128;
    float hid[H1];
#pragma unroll
    for (int j = 0; j < H1; ++j) hid[j] = b1[j];
    for (int k = 0; k < 128; ++k) {
        float x = hr[k];
#pragma unroll
        for (int j = 0; j < H1; ++j) hid[j] += x * W1[k * H1 + j];
    }
    float acc = b2[0];
#pragma unroll
    for (int j = 0; j < H1; ++j) {
        float v = hid[j];
        v = v > 0.0f ? v : LEAKY * v;
        acc += v * W2[j];
    }
    s[i] = acc > 0.0f ? acc : LEAKY * acc;
}

__global__ void k_minmax(const float* __restrict__ s, float* __restrict__ bmax,
                         float* __restrict__ bmin, int ng)
{
    int g = blockIdx.x * 256 + threadIdx.x;
    if (g >= ng) return;
    const float* p = s + (long)g * NPG;
    float mx = -1e30f, mn = 1e30f;
    for (int i = 0; i < NPG; ++i) { float v = p[i]; mx = fmaxf(mx, v); mn = fminf(mn, v); }
    bmax[g] = mx; bmin[g] = mn;
}

__global__ void k_probs(const float* __restrict__ s, const float* __restrict__ bmax,
                        const float* __restrict__ bmin, float* __restrict__ probs,
                        float* __restrict__ out, int n)
{
    int i = blockIdx.x * 256 + threadIdx.x;
    if (i >= n) return;
    int g = i / NPG;
    float v = (s[i] - bmin[g]) / (bmax[g] + 1e-6f - bmin[g]);
    probs[i] = v;
    out[i] = v;
}

// per-edge terms: pairwise sum and gamma accumulation
__global__ void k_edge_terms(const float* __restrict__ probs, const int* __restrict__ row,
                             const int* __restrict__ col, const float* __restrict__ ea,
                             const int* __restrict__ ebi, float* __restrict__ pairsum,
                             float* __restrict__ gam, int ne)
{
    int e = blockIdx.x * 256 + threadIdx.x;
    if (e >= ne) return;
    float d = probs[row[e]] - probs[col[e]];
    int g = ebi[e];
    atomicAdd(&pairsum[g], d * d * ea[e]);
    atomicAdd(&gam[g], ea[e]);
}

__global__ void k_build_adj(const int* __restrict__ row, const int* __restrict__ col,
                            const int* __restrict__ ebi, float* __restrict__ adj, int ne)
{
    int e = blockIdx.x * 256 + threadIdx.x;
    if (e >= ne) return;
    int g = ebi[e];
    int rl = row[e] - g * NPG;
    int cl = col[e] - g * NPG;
    atomicAdd(&adj[(long)g * (NPG * NPG) + rl * NPG + cl], 1.0f);
}

// ---------------------------------------------------------------------------
// Batched symmetric eigensolver: one block per graph, cyclic parallel Jacobi
// on the 75x75 Laplacian (padded to 76 with 1e9 diagonal; the pad plane stays
// exactly decoupled). mode==0: lap0 from adj; mode==1: lap1 from
// (1-p_i-p_j)^2 * adj. Outputs the 3 smallest eigenvalues per graph.
// ---------------------------------------------------------------------------
#define NP 76
__global__ __launch_bounds__(256) void k_eig(const float* __restrict__ adj,
                                             const float* __restrict__ probs,
                                             float* __restrict__ evout, int mode)
{
    __shared__ float M[NP][NP];
    __shared__ float cs[NP / 2], sn[NP / 2];
    __shared__ int perm[NP];

    const int g = blockIdx.x;
    const int tid = threadIdx.x;
    const float* A = adj + (long)g * (NPG * NPG);
    const float* p = probs + (long)g * NPG;

    for (int i = tid; i < NP * NP; i += 256) {
        int r = i / NP, c = i % NP;
        float v;
        if (r < NPG && c < NPG) {
            float a = A[r * NPG + c];
            if (mode) { float t = 1.0f - p[r] - p[c]; a *= t * t; }
            v = -a;                               // off-diagonal of Laplacian
        } else {
            v = (r == c) ? 1e9f : 0.0f;           // decoupled pad
        }
        M[r][c] = v;
    }
    if (tid < NP) perm[tid] = tid;
    __syncthreads();
    if (tid < NPG) {                              // diag: rowsum - a_ii (M has -a_ii)
        float rs = 0.0f;
        for (int c = 0; c < NPG; ++c) {
            float a = A[tid * NPG + c];
            if (mode) { float t = 1.0f - p[tid] - p[c]; a *= t * t; }
            rs += a;
        }
        M[tid][tid] += rs;
    }
    __syncthreads();

    const int SWEEPS = 14;
    for (int sw = 0; sw < SWEEPS; ++sw) {
        for (int round = 0; round < NP - 1; ++round) {
            // phase A: rotation angles for the 38 disjoint pairs
            if (tid < NP / 2) {
                int pp = perm[tid], qq = perm[NP - 1 - tid];
                if (pp > qq) { int t = pp; pp = qq; qq = t; }
                float app = M[pp][pp], aqq = M[qq][qq], apq = M[pp][qq];
                float c = 1.0f, s = 0.0f;
                if (fabsf(apq) > 1e-12f) {
                    float tau = (aqq - app) / (2.0f * apq);
                    float t = (tau >= 0.0f) ? 1.0f / (tau + sqrtf(1.0f + tau * tau))
                                            : 1.0f / (tau - sqrtf(1.0f + tau * tau));
                    c = rsqrtf(1.0f + t * t);
                    s = t * c;
                }
                cs[tid] = c; sn[tid] = s;
            }
            __syncthreads();
            // phase B: M <- M * J   (column rotations; disjoint columns per pair)
            for (int w = tid; w < (NP / 2) * NP; w += 256) {
                int i = w / NP, r = w % NP;
                int pp = perm[i], qq = perm[NP - 1 - i];
                if (pp > qq) { int t = pp; pp = qq; qq = t; }
                float c = cs[i], s = sn[i];
                float mp = M[r][pp], mq = M[r][qq];
                M[r][pp] = c * mp - s * mq;
                M[r][qq] = s * mp + c * mq;
            }
            __syncthreads();
            // phase C: M <- J^T * M  (row rotations)
            for (int w = tid; w < (NP / 2) * NP; w += 256) {
                int i = w / NP, k = w % NP;
                int pp = perm[i], qq = perm[NP - 1 - i];
                if (pp > qq) { int t = pp; pp = qq; qq = t; }
                float c = cs[i], s = sn[i];
                float mp = M[pp][k], mq = M[qq][k];
                M[pp][k] = c * mp - s * mq;
                M[qq][k] = s * mp + c * mq;
            }
            __syncthreads();
            // round-robin rotation of the tournament schedule
            if (tid == 0) {
                int last = perm[NP - 1];
                for (int i = NP - 1; i > 1; --i) perm[i] = perm[i - 1];
                perm[1] = last;
            }
            __syncthreads();
        }
    }

    if (tid == 0) {
        float e0 = 1e30f, e1 = 1e30f, e2 = 1e30f;
        for (int i = 0; i < NPG; ++i) {
            float d = M[i][i];
            if (d < e0)      { e2 = e1; e1 = e0; e0 = d; }
            else if (d < e1) { e2 = e1; e1 = d; }
            else if (d < e2) { e2 = d; }
        }
        evout[g * 3 + 0] = e0; evout[g * 3 + 1] = e1; evout[g * 3 + 2] = e2;
    }
}

__global__ void k_loss(const float* __restrict__ ev0, const float* __restrict__ ev1,
                       const float* __restrict__ pairsum, const float* __restrict__ gam,
                       float* __restrict__ loss_out, float* __restrict__ f_out,
                       float* __restrict__ g_out, int ng)
{
    int g = blockIdx.x * 256 + threadIdx.x;
    if (g >= ng) return;
    float aaa   = logf(1e-4f) / (-ev0[g * 3 + 2]);
    float lam3  = ev1[g * 3 + 2] - ev1[g * 3 + 1];
    float gamma = gam[g] * 0.5f;
    float fr    = gamma - pairsum[g] * 0.5f;
    float gr    = expf(-aaa * lam3);
    float alpha = (2.0f * EH) / 72.0f;        // edges_per_graph / (2*36)
    loss_out[g] = alpha * fr + gamma * gr;
    f_out[g] = fr;
    g_out[g] = gr;
}

__global__ __launch_bounds__(256) void k_mean(const float* __restrict__ loss,
                                              float* __restrict__ out, int ng)
{
    __shared__ float red[256];
    float s = 0.0f;
    for (int i = threadIdx.x; i < ng; i += 256) s += loss[i];
    red[threadIdx.x] = s;
    __syncthreads();
    for (int off = 128; off > 0; off >>= 1) {
        if (threadIdx.x < off) red[threadIdx.x] += red[threadIdx.x + off];
        __syncthreads();
    }
    if (threadIdx.x == 0) out[0] = red[0] / (float)ng;
}

// ---------------------------------------------------------------------------
extern "C" void kernel_launch(void* const* d_in, const int* in_sizes, int n_in,
                              void* d_out, int out_size, void* d_ws, size_t ws_size,
                              hipStream_t stream)
{
    (void)in_sizes; (void)n_in; (void)out_size; (void)ws_size;

    const float* x          = (const float*)d_in[0];
    const float* ea         = (const float*)d_in[1];
    const float* conv1_eps  = (const float*)d_in[2];
    const float* conv1_W1   = (const float*)d_in[3];
    const float* conv1_b1   = (const float*)d_in[4];
    const float* conv1_W2   = (const float*)d_in[5];
    const float* conv1_b2   = (const float*)d_in[6];
    const float* conv1_bn_g = (const float*)d_in[7];
    const float* conv1_bn_b = (const float*)d_in[8];
    const float* bn1_g      = (const float*)d_in[9];
    const float* bn1_b      = (const float*)d_in[10];
    const float* convs_eps  = (const float*)d_in[11];
    const float* convs_W1   = (const float*)d_in[12];
    const float* convs_b1   = (const float*)d_in[13];
    const float* convs_W2   = (const float*)d_in[14];
    const float* convs_b2   = (const float*)d_in[15];
    const float* convs_bn_g = (const float*)d_in[16];
    const float* convs_bn_b = (const float*)d_in[17];
    const float* bns_g      = (const float*)d_in[18];
    const float* bns_b      = (const float*)d_in[19];
    const float* lin1_W     = (const float*)d_in[20];
    const float* lin1_b     = (const float*)d_in[21];
    const float* lin2_W     = (const float*)d_in[22];
    const float* lin2_b     = (const float*)d_in[23];
    const int*   eidx       = (const int*)d_in[24];
    const int*   ebi        = (const int*)d_in[26];
    const int*   row        = eidx;
    const int*   col        = eidx + EE;

    float* out = (float*)d_out;

    // workspace layout (floats)
    float* W = (float*)d_ws;
    size_t o = 0;
    const size_t ND = (size_t)NN * DD;
    float* h      = W + o; o += ND;
    float* zb     = W + o; o += ND;
    float* agg    = W + o; o += ND;
    float* aggs   = W + o; o += NN;
    float* sbuf   = W + o; o += NN;
    float* probs  = W + o; o += NN;
    float* bnsum  = W + o; o += 128;
    float* bnsq   = W + o; o += 128;
    float* bmax   = W + o; o += GG;
    float* bmin   = W + o; o += GG;
    float* pairsum= W + o; o += GG;   // pairsum and gam adjacent: one memset
    float* gam    = W + o; o += GG;
    float* ev0    = W + o; o += (size_t)GG * 3;
    float* ev1    = W + o; o += (size_t)GG * 3;
    _Float16* wpack = (_Float16*)(W + o); o += 8192;   // 16384 halves
    float* adj    = W + o; o += (size_t)GG * NPG * NPG;

    const long  total = (long)ND;
    const int   eBlk  = (EE + 255) / 256;
    const int   ndBlk = (int)((total + 255) / 256);
    const int   avBlk = (int)(((long)EE * 128 + 255) / 256);
    const int   gemmBlk = NN / 16;
    const int   statBlk = (NN + 255) / 256;   // 256 rows per block
    const float invN = 1.0f / (float)NN;

    // ---------------- conv1 ----------------
    hipMemsetAsync(aggs, 0, (size_t)NN * sizeof(float), stream);
    k_agg_scalar<<<eBlk, 256, 0, stream>>>(x, row, col, aggs, EE);
    k_conv1_expand<<<ndBlk, 256, 0, stream>>>(x, aggs, conv1_eps, conv1_W1, conv1_b1, zb, total);
    k_packW<<<64, 256, 0, stream>>>(conv1_W2, wpack);
    k_gemm128_wmma<<<gemmBlk, 256, 0, stream>>>(zb, wpack, conv1_b2, agg);
    hipMemsetAsync(bnsum, 0, 256 * sizeof(float), stream);
    k_bn_stats<<<statBlk, 128, 0, stream>>>(agg, bnsum, bnsq, 256, NN);
    k_bn_apply<<<ndBlk, 256, 0, stream>>>(agg, zb, bnsum, bnsq, conv1_bn_g, conv1_bn_b, 1, total, invN);
    k_resscale<<<ndBlk, 256, 0, stream>>>(h, zb, 0, total);
    hipMemsetAsync(bnsum, 0, 256 * sizeof(float), stream);
    k_bn_stats<<<statBlk, 128, 0, stream>>>(h, bnsum, bnsq, 256, NN);
    k_bn_apply<<<ndBlk, 256, 0, stream>>>(h, h, bnsum, bnsq, bn1_g, bn1_b, 0, total, invN);

    // ---------------- layers 2..4 ----------------
    for (int i = 0; i < NL - 1; ++i) {
        hipMemsetAsync(agg, 0, ND * sizeof(float), stream);
        k_agg_vec<<<avBlk, 256, 0, stream>>>(h, row, col, agg, (long)EE * 128);
        k_axpy_eps<<<ndBlk, 256, 0, stream>>>(agg, h, convs_eps + i, total);
        k_packW<<<64, 256, 0, stream>>>(convs_W1 + (size_t)i * DD * DD, wpack);
        k_gemm128_wmma<<<gemmBlk, 256, 0, stream>>>(agg, wpack, convs_b1 + (size_t)i * DD, zb);
        k_packW<<<64, 256, 0, stream>>>(convs_W2 + (size_t)i * DD * DD, wpack);
        k_gemm128_wmma<<<gemmBlk, 256, 0, stream>>>(zb, wpack, convs_b2 + (size_t)i * DD, agg);
        hipMemsetAsync(bnsum, 0, 256 * sizeof(float), stream);
        k_bn_stats<<<statBlk, 128, 0, stream>>>(agg, bnsum, bnsq, 256, NN);
        k_bn_apply<<<ndBlk, 256, 0, stream>>>(agg, zb, bnsum, bnsq,
                                              convs_bn_g + (size_t)i * DD,
                                              convs_bn_b + (size_t)i * DD, 1, total, invN);
        k_resscale<<<ndBlk, 256, 0, stream>>>(h, zb, 1, total);
        hipMemsetAsync(bnsum, 0, 256 * sizeof(float), stream);
        k_bn_stats<<<statBlk, 128, 0, stream>>>(h, bnsum, bnsq, 256, NN);
        k_bn_apply<<<ndBlk, 256, 0, stream>>>(h, h, bnsum, bnsq,
                                              bns_g + (size_t)i * DD,
                                              bns_b + (size_t)i * DD, 0, total, invN);
    }

    // ---------------- head + probs ----------------
    k_lin<<<(NN + 255) / 256, 256, 0, stream>>>(h, lin1_W, lin1_b, lin2_W, lin2_b, sbuf, NN);
    k_minmax<<<(GG + 255) / 256, 256, 0, stream>>>(sbuf, bmax, bmin, GG);
    k_probs<<<(NN + 255) / 256, 256, 0, stream>>>(sbuf, bmax, bmin, probs, out, NN);

    // ---------------- edge terms + adjacency ----------------
    hipMemsetAsync(pairsum, 0, (size_t)2 * GG * sizeof(float), stream);
    hipMemsetAsync(adj, 0, (size_t)GG * NPG * NPG * sizeof(float), stream);
    k_edge_terms<<<eBlk, 256, 0, stream>>>(probs, row, col, ea, ebi, pairsum, gam, EE);
    k_build_adj<<<eBlk, 256, 0, stream>>>(row, col, ebi, adj, EE);

    // ---------------- eigensolves ----------------
    k_eig<<<GG, 256, 0, stream>>>(adj, probs, ev0, 0);
    k_eig<<<GG, 256, 0, stream>>>(adj, probs, ev1, 1);

    // ---------------- loss + mean ----------------
    float* loss_out = out + NN;
    float* mean_out = out + NN + GG;
    float* f_out    = out + NN + GG + 1;
    float* g_out    = out + NN + 2 * GG + 1;
    k_loss<<<(GG + 255) / 256, 256, 0, stream>>>(ev0, ev1, pairsum, gam, loss_out, f_out, g_out, GG);
    k_mean<<<1, 256, 0, stream>>>(loss_out, mean_out, GG);
}
